// multihead_self_attention_58798102282657
// MI455X (gfx1250) — matrix operand order repaired and checked by reading, not compiled, not run on hardware
//
#include <hip/hip_runtime.h>
#include <hip/hip_bf16.h>

// ---------------------------------------------------------------------------
// MI455X (gfx1250) fused causal MHA with RoPE.
//   K0: one-pass f32->f16 conversion of x and the four weight matrices
//   K1: X*W^T GEMM (pure-f16 operands, f32 WMMA accum) + RoPE
//       -> Q,K [b][h][s][dk], V^T [b][h][dk][s]
//   K2: flash attention, 16 q-rows/wave, 32 keys/iter, online softmax
//   K3: attn * W_O^T -> f32 output
// All matrix math via V_WMMA_F32_16X16X32_F16 (wave32 layouts per CDNA5 ISA
// 7.12.2).  Pre-converting operands to f16 keeps the GEMM inner loops free of
// v_cvt VALU (which otherwise matches the WMMA issue rate and forces
// WMMA->VALU WAR hazard nops) and halves fragment load bytes.
// ---------------------------------------------------------------------------

#define SEQL   4096
#define DMODEL 768
#define NHEAD  12
#define DKH    64

typedef __attribute__((ext_vector_type(16))) _Float16 v16h;
typedef __attribute__((ext_vector_type(8)))  float    v8f;
typedef __attribute__((ext_vector_type(4)))  float    f32x4;
typedef __attribute__((ext_vector_type(4)))  unsigned int u32x4;

union AB16 {           // one wave32 A/B fragment for 16x16x32 f16 WMMA
  v16h v;
  u32x4 q[2];          // two 16-byte chunks
  _Float16 e[16];
};

__device__ __forceinline__ v8f wmma16(v16h a, v16h b, v8f c) {
  return __builtin_amdgcn_wmma_f32_16x16x32_f16(false, a, false, b, (short)0, c,
                                                false, false);
}

// ---------------------------------------------------------------------------
// Kernel 0: bulk f32 -> f16 conversion (8 elements / thread).
// ---------------------------------------------------------------------------
__global__ void __launch_bounds__(256) cvt_f32_f16_kernel(
    const float* __restrict__ in, _Float16* __restrict__ out, int n8)
{
  const int idx = blockIdx.x * 256 + threadIdx.x;
  if (idx >= n8) return;
  const f32x4* p = (const f32x4*)(in + (size_t)idx * 8);
  const f32x4 a = p[0], b = p[1];
  union { u32x4 q; _Float16 e[8]; } o;
#pragma unroll
  for (int j = 0; j < 4; ++j) {
    o.e[j]     = (_Float16)a[j];
    o.e[4 + j] = (_Float16)b[j];
  }
  *(u32x4*)(out + (size_t)idx * 8) = o.q;
}

// ---------------------------------------------------------------------------
// Kernel 1: projections + RoPE (f16 operands).
//   mode 0/1: out[b][h][s][dk] (f16) with RoPE applied (Q or K)
//   mode 2  : out[b][h][dk][s] (f16) transposed V (no RoPE)
// Wave computes a 16-row x 128-col tile; block = 8 waves = 128 rows.
// grid = (8192/128, 768/128) = (64, 6)
// ---------------------------------------------------------------------------
__global__ void __launch_bounds__(256) qkv_rope_kernel(
    const _Float16* __restrict__ xh, const int* __restrict__ tpos,
    const _Float16* __restrict__ W, _Float16* __restrict__ outp, int mode)
{
  const int lane  = threadIdx.x & 31;
  const int wave  = threadIdx.x >> 5;
  const int laneN = lane & 15;
  const int hiOff = (lane & 16) ? 8 : 0;   // C-row offset
  const int kbB   = (lane & 16) ? 16 : 0;  // B-frag K base
  const int rowbase = blockIdx.x * 128 + wave * 16;
  const int colbase = blockIdx.y * 128;

  const _Float16* arow = xh + (size_t)(rowbase + laneN) * DMODEL;

  v8f acc[8] = {};

  for (int ks = 0; ks < DMODEL / 32; ++ks) {
    // A fragment: row m = laneN, 16 contiguous K halves [kb, kb+16)
    AB16 a;
    const _Float16* pa = arow + ks * 32 + kbB;
    a.q[0] = *(const u32x4*)(pa);
    a.q[1] = *(const u32x4*)(pa + 8);
#pragma unroll
    for (int ct = 0; ct < 8; ++ct) {
      // B fragment: col n = W row (B = W^T), 16 contiguous K halves
      const _Float16* pb =
          W + (size_t)(colbase + ct * 16 + laneN) * DMODEL + ks * 32 + kbB;
      AB16 bb;
      bb.q[0] = *(const u32x4*)(pb);
      bb.q[1] = *(const u32x4*)(pb + 8);
      acc[ct] = wmma16(a.v, bb.v, acc[ct]);
    }
  }

  if (mode < 2) {
    // RoPE + store [b][h][s][dk].  Pair partner (dk even<->odd) sits in lane^1.
    float posf[8];
#pragma unroll
    for (int i = 0; i < 8; ++i) posf[i] = (float)tpos[rowbase + hiOff + i];
#pragma unroll
    for (int ct = 0; ct < 8; ++ct) {
      const int col = colbase + ct * 16 + laneN;
      const int h = col >> 6, dki = col & 63;
      // theta^{-2p/dk} = exp(-ln(1e4)/32 * p)
      const float invf = __expf(-0.28782313662425572f * (float)(dki >> 1));
      const bool odd = (dki & 1) != 0;
#pragma unroll
      for (int i = 0; i < 8; ++i) {
        const float v = acc[ct][i];
        const float partner = __shfl_xor(v, 1, 32);
        float sn, cs;
        __sincosf(posf[i] * invf, &sn, &cs);
        const float r = odd ? (sn * partner + cs * v) : (cs * v - sn * partner);
        const int row = rowbase + hiOff + i;
        const int bb_ = row >> 12, sidx = row & (SEQL - 1);
        outp[((size_t)(bb_ * NHEAD + h) * SEQL + sidx) * DKH + dki] =
            (_Float16)r;
      }
    }
  } else {
    // V transposed: [b][h][dk][s]
#pragma unroll
    for (int ct = 0; ct < 8; ++ct) {
      const int col = colbase + ct * 16 + laneN;
      const int h = col >> 6, dki = col & 63;
#pragma unroll
      for (int i = 0; i < 8; ++i) {
        const int row = rowbase + hiOff + i;
        const int bb_ = row >> 12, sidx = row & (SEQL - 1);
        outp[((size_t)(bb_ * NHEAD + h) * DKH + dki) * SEQL + sidx] =
            (_Float16)acc[ct][i];
      }
    }
  }
}

// ---------------------------------------------------------------------------
// Kernel 2: causal flash attention.
// grid = (4096/64, 12, 2), block = 128 (4 waves); wave owns 16 query rows.
// Per iteration: 32 keys -> 4 QK^T WMMAs, online softmax, P->LDS transpose,
// 4 PV WMMAs against V^T.
// ---------------------------------------------------------------------------
__global__ void __launch_bounds__(128) flash_attn_kernel(
    const _Float16* __restrict__ Q, const _Float16* __restrict__ K,
    const _Float16* __restrict__ VT, _Float16* __restrict__ attn)
{
  __shared__ __align__(16) _Float16 psh[4][16 * 40];  // stride 40 halves (80B)
  const int lane  = threadIdx.x & 31;
  const int wave  = threadIdx.x >> 5;
  const int laneN = lane & 15;
  const int hiOff = (lane & 16) ? 8 : 0;
  const int kbB   = (lane & 16) ? 16 : 0;
  const int b = blockIdx.z, h = blockIdx.y;
  const int qr0 = blockIdx.x * 64 + wave * 16;
  const size_t base = (size_t)(b * NHEAD + h) * SEQL * DKH;  // Q/K and VT match

  // Q fragments (reused across all key tiles): K-dim = dk = 64 -> 2 frags
  const _Float16* qrow = Q + base + (size_t)(qr0 + laneN) * DKH;
  AB16 aq0, aq1;
  aq0.q[0] = *(const u32x4*)(qrow + kbB);
  aq0.q[1] = *(const u32x4*)(qrow + kbB + 8);
  aq1.q[0] = *(const u32x4*)(qrow + 32 + kbB);
  aq1.q[1] = *(const u32x4*)(qrow + 32 + kbB + 8);

  v8f o[4] = {};
  float mrow[8], lrow[8];
#pragma unroll
  for (int i = 0; i < 8; ++i) { mrow[i] = -3.0e38f; lrow[i] = 0.0f; }

  _Float16* pw = &psh[wave][0];
  const int ktmax = (qr0 + 15) >> 5;

  for (int kt = 0; kt <= ktmax; ++kt) {
    const int kb0 = kt << 5;
    const _Float16* kr0 = K + base + (size_t)(kb0 + laneN) * DKH;
    const _Float16* kr1 = kr0 + 16 * DKH;

    v8f s0 = {}, s1 = {};
    AB16 bk;
    bk.q[0] = *(const u32x4*)(kr0 + kbB);
    bk.q[1] = *(const u32x4*)(kr0 + kbB + 8);
    s0 = wmma16(aq0.v, bk.v, s0);
    bk.q[0] = *(const u32x4*)(kr0 + 32 + kbB);
    bk.q[1] = *(const u32x4*)(kr0 + 32 + kbB + 8);
    s0 = wmma16(aq1.v, bk.v, s0);
    bk.q[0] = *(const u32x4*)(kr1 + kbB);
    bk.q[1] = *(const u32x4*)(kr1 + kbB + 8);
    s1 = wmma16(aq0.v, bk.v, s1);
    bk.q[0] = *(const u32x4*)(kr1 + 32 + kbB);
    bk.q[1] = *(const u32x4*)(kr1 + 32 + kbB + 8);
    s1 = wmma16(aq1.v, bk.v, s1);

    const bool needMask = (kb0 + 31) > qr0;  // wave-uniform
    float pscale[8];
#pragma unroll
    for (int i = 0; i < 8; ++i) {
      const int qg = qr0 + hiOff + i;
      float v0 = s0[i] * 0.125f;   // 1/sqrt(dk)
      float v1 = s1[i] * 0.125f;
      if (needMask) {
        if (kb0 + laneN      > qg) v0 = -3.0e38f;
        if (kb0 + 16 + laneN > qg) v1 = -3.0e38f;
      }
      float mx = fmaxf(v0, v1);
#pragma unroll
      for (int off = 1; off < 16; off <<= 1)
        mx = fmaxf(mx, __shfl_xor(mx, off, 32));
      const float mnew = fmaxf(mrow[i], mx);
      const float sc = __expf(mrow[i] - mnew);
      const float p0 = __expf(v0 - mnew);
      const float p1 = __expf(v1 - mnew);
      float rs = p0 + p1;
#pragma unroll
      for (int off = 1; off < 16; off <<= 1) rs += __shfl_xor(rs, off, 32);
      lrow[i] = lrow[i] * sc + rs;
      mrow[i] = mnew;
      pscale[i] = sc;
      s0[i] = p0;
      s1[i] = p1;
    }
#pragma unroll
    for (int c = 0; c < 4; ++c)
#pragma unroll
      for (int i = 0; i < 8; ++i) o[c][i] *= pscale[i];

    // C-layout P -> row-major LDS tile -> A-layout fragment (same-wave DS order)
#pragma unroll
    for (int i = 0; i < 8; ++i) {
      const int M = hiOff + i;
      pw[M * 40 + laneN]      = (_Float16)s0[i];
      pw[M * 40 + 16 + laneN] = (_Float16)s1[i];
    }
    AB16 pa;
    const _Float16* pr = pw + laneN * 40 + kbB;
    pa.q[0] = *(const u32x4*)(pr);
    pa.q[1] = *(const u32x4*)(pr + 8);

#pragma unroll
    for (int c = 0; c < 4; ++c) {
      const _Float16* vp =
          VT + base + (size_t)(c * 16 + laneN) * SEQL + kb0 + kbB;
      AB16 bv;
      bv.q[0] = *(const u32x4*)(vp);
      bv.q[1] = *(const u32x4*)(vp + 8);
      o[c] = wmma16(pa.v, bv.v, o[c]);
    }
  }

#pragma unroll
  for (int i = 0; i < 8; ++i) lrow[i] = 1.0f / lrow[i];
#pragma unroll
  for (int c = 0; c < 4; ++c)
#pragma unroll
    for (int i = 0; i < 8; ++i) {
      const int sidx = qr0 + hiOff + i;
      attn[((size_t)b * SEQL + sidx) * DMODEL + h * DKH + c * 16 + laneN] =
          (_Float16)(o[c][i] * lrow[i]);
    }
}

// ---------------------------------------------------------------------------
// Kernel 3: output projection attn[8192x768](f16) * W_O^T(f16) -> out(f32)
// grid = (64, 6), block = 256; wave computes 16 x 128 tile.
// ---------------------------------------------------------------------------
__global__ void __launch_bounds__(256) oproj_kernel(
    const _Float16* __restrict__ attn, const _Float16* __restrict__ WO,
    float* __restrict__ outp)
{
  const int lane  = threadIdx.x & 31;
  const int wave  = threadIdx.x >> 5;
  const int laneN = lane & 15;
  const int hiOff = (lane & 16) ? 8 : 0;
  const int kbB   = (lane & 16) ? 16 : 0;
  const int rowbase = blockIdx.x * 128 + wave * 16;
  const int colbase = blockIdx.y * 128;
  const _Float16* arow = attn + (size_t)(rowbase + laneN) * DMODEL;

  v8f acc[8] = {};

  for (int ks = 0; ks < DMODEL / 32; ++ks) {
    AB16 a;
    const _Float16* pa = arow + ks * 32 + kbB;
    a.q[0] = *(const u32x4*)(pa);
    a.q[1] = *(const u32x4*)(pa + 8);
#pragma unroll
    for (int ct = 0; ct < 8; ++ct) {
      const _Float16* pb =
          WO + (size_t)(colbase + ct * 16 + laneN) * DMODEL + ks * 32 + kbB;
      AB16 bb;
      bb.q[0] = *(const u32x4*)(pb);
      bb.q[1] = *(const u32x4*)(pb + 8);
      acc[ct] = wmma16(a.v, bb.v, acc[ct]);
    }
  }
#pragma unroll
  for (int ct = 0; ct < 8; ++ct)
#pragma unroll
    for (int i = 0; i < 8; ++i)
      outp[(size_t)(rowbase + hiOff + i) * DMODEL + colbase + ct * 16 + laneN] =
          acc[ct][i];
}

// ---------------------------------------------------------------------------
extern "C" void kernel_launch(void* const* d_in, const int* in_sizes, int n_in,
                              void* d_out, int out_size, void* d_ws,
                              size_t ws_size, hipStream_t stream) {
  (void)in_sizes; (void)n_in; (void)out_size; (void)ws_size;
  const float* x    = (const float*)d_in[0];
  const int*   tpos = (const int*)d_in[1];
  const float* WQ   = (const float*)d_in[2];
  const float* WK   = (const float*)d_in[3];
  const float* WV   = (const float*)d_in[4];
  const float* WO   = (const float*)d_in[5];
  float* out = (float*)d_out;

  // Workspace layout (all f16 halves):
  //   Q, K ([b][h][s][dk]), V^T ([b][h][dk][s]), attn ([b][s][768]),
  //   x_f16 ([8192][768]), then 4 weight matrices ([768][768]).
  const size_t MAT = (size_t)2 * NHEAD * SEQL * DKH;  // 6291456
  const size_t WSZ = (size_t)DMODEL * DMODEL;         // 589824
  _Float16* ws  = (_Float16*)d_ws;
  _Float16* Qb  = ws;
  _Float16* Kb  = Qb + MAT;
  _Float16* VTb = Kb + MAT;
  _Float16* Ab  = VTb + MAT;
  _Float16* Xh  = Ab + MAT;
  _Float16* WQh = Xh + MAT;
  _Float16* WKh = WQh + WSZ;
  _Float16* WVh = WKh + WSZ;
  _Float16* WOh = WVh + WSZ;

  // K0: one-pass conversions
  const int n8x = (int)(MAT / 8);
  const int n8w = (int)(WSZ / 8);
  cvt_f32_f16_kernel<<<(n8x + 255) / 256, 256, 0, stream>>>(x, Xh, n8x);
  cvt_f32_f16_kernel<<<(n8w + 255) / 256, 256, 0, stream>>>(WQ, WQh, n8w);
  cvt_f32_f16_kernel<<<(n8w + 255) / 256, 256, 0, stream>>>(WK, WKh, n8w);
  cvt_f32_f16_kernel<<<(n8w + 255) / 256, 256, 0, stream>>>(WV, WVh, n8w);
  cvt_f32_f16_kernel<<<(n8w + 255) / 256, 256, 0, stream>>>(WO, WOh, n8w);

  dim3 gProj(64, 6, 1);
  qkv_rope_kernel<<<gProj, 256, 0, stream>>>(Xh, tpos, WQh, Qb, 0);
  qkv_rope_kernel<<<gProj, 256, 0, stream>>>(Xh, tpos, WKh, Kb, 1);
  qkv_rope_kernel<<<gProj, 256, 0, stream>>>(Xh, tpos, WVh, VTb, 2);

  flash_attn_kernel<<<dim3(SEQL / 64, NHEAD, 2), 128, 0, stream>>>(Qb, Kb, VTb,
                                                                   Ab);

  oproj_kernel<<<gProj, 256, 0, stream>>>(Ab, WOh, out);
}
